// Mlp_StaticRoutedLoRAExpert_46737834115830
// MI455X (gfx1250) — compile-verified, matrix operand order checked by reading, not run on hardware
//
#include <hip/hip_runtime.h>
#include <hip/hip_bf16.h>
#include <cstdint>
#include <cstddef>

// ---------------------------------------------------------------------------
// Problem constants (match reference): B=32,S=1280,IN=768,HID=3072,OUT=768,
// E=2,R=64, SCALE = 128/64 = 2.0
// ---------------------------------------------------------------------------
static constexpr int   Bq   = 32;
static constexpr int   Sq   = 1280;
static constexpr int   INq  = 768;
static constexpr int   HIDq = 3072;
static constexpr int   OUTq = 768;
static constexpr int   Eq   = 2;
static constexpr int   Rq   = 64;
static constexpr int   Mq   = Bq * Sq;          // 40960 tokens
static constexpr float SCALE_LORA = 2.0f;

typedef __bf16 bf16_t;
typedef __attribute__((ext_vector_type(16))) __bf16 v16bf;
typedef __attribute__((ext_vector_type(8)))  __bf16 v8bf;
typedef __attribute__((ext_vector_type(4)))  __bf16 v4bf;
typedef __attribute__((ext_vector_type(8)))  float  v8f;

// GEMM tiling: block tile 128(M) x 256(N) x 32(K); 8 waves, each 64x64.
static constexpr int BM  = 128;
static constexpr int BN  = 256;
static constexpr int BK  = 32;   // one v_wmma_f32_16x16x32_bf16 step
static constexpr int LDT = 40;   // LDS row stride (elems): 80B -> conflict-free b128

// ---------------------------------------------------------------------------
// CDNA5 async global->LDS copy (ASYNCcnt-tracked DMA into LDS).
// Generic LDS pointer low 32 bits == LDS byte address (aperture truncation).
// ---------------------------------------------------------------------------
__device__ __forceinline__ void async_cp16(void* lds, const void* gaddr)
{
    uint32_t loff = (uint32_t)(uintptr_t)lds;
    asm volatile("global_load_async_to_lds_b128 %0, %1, off"
                 :: "v"(loff), "v"(gaddr)
                 : "memory");
}
__device__ __forceinline__ void wait_async0()
{
    asm volatile("s_wait_asynccnt 0" ::: "memory");
}

// ---------------------------------------------------------------------------
// x (f32) -> bf16, 4 elements / thread
// ---------------------------------------------------------------------------
__global__ __launch_bounds__(256) void f32_to_bf16_kernel(
    const float* __restrict__ in, bf16_t* __restrict__ out, size_t n4)
{
    size_t i = (size_t)blockIdx.x * blockDim.x + threadIdx.x;
    if (i >= n4) return;
    float4 f = ((const float4*)in)[i];
    v4bf o;
    o.x = (bf16_t)f.x; o.y = (bf16_t)f.y; o.z = (bf16_t)f.z; o.w = (bf16_t)f.w;
    *(v4bf*)(out + i * 4) = o;
}

// ---------------------------------------------------------------------------
// Merge LoRA into dense weight:  Wm[e][n][k] = w[n][k] + SCALE * sum_r b[e][n][r]*a[e][r][k]
// one block per (n, e); b-row cached in LDS; k coalesced across threads.
// ---------------------------------------------------------------------------
__global__ __launch_bounds__(256) void merge_lora_kernel(
    const float* __restrict__ w, const float* __restrict__ a,
    const float* __restrict__ b, bf16_t* __restrict__ out,
    int N, int K, int R)
{
    const int n = blockIdx.x;
    const int e = blockIdx.y;
    __shared__ float bv[64];
    if ((int)threadIdx.x < R)
        bv[threadIdx.x] = b[((size_t)e * N + n) * R + threadIdx.x];
    __syncthreads();
    const float* arow = a + (size_t)e * R * K;
    bf16_t* orow = out + ((size_t)e * N + n) * K;
    const float* wrow = w + (size_t)n * K;
    for (int k = threadIdx.x; k < K; k += 256) {
        float s = 0.0f;
        #pragma unroll 8
        for (int r = 0; r < R; ++r)
            s += bv[r] * arow[(size_t)r * K + k];
        float v = wrow[k] + SCALE_LORA * s;
        orow[k] = (bf16_t)v;
    }
}

// ---------------------------------------------------------------------------
// bf16 WMMA GEMM:  C[M,N] = A[M,K] * W_e[N,K]^T + bias  (+ exact GeLU)
// 256 threads = 8 waves (2x4); per wave 64x64 C-slab (4x4 wmma frags).
// Double-buffered LDS fed by async global->LDS DMA, overlapped with WMMA.
// Expert picked per 128-row tile from (chunk_sizes, expert_indices).
// ---------------------------------------------------------------------------
template<bool GELU, bool OUT_BF16>
__global__ __launch_bounds__(256) void gemm_bf16_wmma(
    const bf16_t* __restrict__ A, const bf16_t* __restrict__ Wbase,
    const float*  __restrict__ bias, void* __restrict__ Cout,
    int M, int N, int K, int S,
    const long long* __restrict__ expert_indices,
    const long long* __restrict__ chunk_sizes, int n_chunks)
{
    __shared__ __align__(16) bf16_t Alds[2][BM * LDT];
    __shared__ __align__(16) bf16_t Blds[2][BN * LDT];

    const int tid   = threadIdx.x;
    const int lane  = tid & 31;
    const int wave  = tid >> 5;
    const int wm    = wave & 1;    // 0..1 -> 64-row slab
    const int wn    = wave >> 1;   // 0..3 -> 64-col slab
    const int lmod  = lane & 15;
    const int lhalf = lane >> 4;

    const int tileN = blockIdx.x * BN;
    const int tileM = blockIdx.y * BM;

    // --- expert selection for this token tile (tile never crosses a chunk) ---
    long long s0 = (long long)tileM % S;
    long long accsz = 0, e = 0;
    for (int c = 0; c < n_chunks; ++c) {
        accsz += chunk_sizes[c];
        if (s0 < accsz) { e = expert_indices[c]; break; }
    }
    const bf16_t* Wgt = Wbase + (size_t)e * (size_t)N * (size_t)K;

    // --- tile staging: each thread DMAs 2x16B (A) + 4x16B (B) per K-stage ---
    const int lrow = tid >> 2;          // 0..63
    const int lcol = (tid & 3) * 8;     // elems (16B granules)
    const bf16_t* gA = A   + (size_t)(tileM + lrow) * K + lcol;
    const bf16_t* gB = Wgt + (size_t)(tileN + lrow) * K + lcol;

    auto stage = [&](int buf, int k0) {
        async_cp16(&Alds[buf][(lrow      ) * LDT + lcol], gA + k0);
        async_cp16(&Alds[buf][(lrow +  64) * LDT + lcol], gA + (size_t)64 * K + k0);
        #pragma unroll
        for (int j = 0; j < 4; ++j)
            async_cp16(&Blds[buf][(lrow + 64 * j) * LDT + lcol],
                       gB + (size_t)(64 * j) * K + k0);
    };
    // 16-bit A/B fragment (16x32 / 32x16): lane holds row lmod, K-halves by lhalf
    auto frag = [&](const bf16_t* base, int row0) -> v16bf {
        const bf16_t* p = base + (size_t)(row0 + lmod) * LDT + lhalf * 8;
        v8bf lo = *(const v8bf*)p;
        v8bf hi = *(const v8bf*)(p + 16);
        return __builtin_shufflevector(lo, hi, 0,1,2,3,4,5,6,7,8,9,10,11,12,13,14,15);
    };

    v8f accf[4][4] = {};
    const int nk = K / BK;

    stage(0, 0);
    wait_async0();
    __syncthreads();

    for (int ks = 0; ks < nk; ++ks) {
        const int cur = ks & 1;
        if (ks + 1 < nk) stage(cur ^ 1, (ks + 1) * BK);   // DMA next stage

        v16bf bfr[4];
        #pragma unroll
        for (int ni = 0; ni < 4; ++ni)
            bfr[ni] = frag(&Blds[cur][0], wn * 64 + ni * 16);
        #pragma unroll
        for (int mi = 0; mi < 4; ++mi) {
            v16bf afr = frag(&Alds[cur][0], wm * 64 + mi * 16);
            #pragma unroll
            for (int ni = 0; ni < 4; ++ni)
                accf[mi][ni] = __builtin_amdgcn_wmma_f32_16x16x32_bf16(
                    false, afr, false, bfr[ni], (short)0, accf[mi][ni], false, false);
        }

        if (ks + 1 < nk) wait_async0();                   // next stage landed
        __syncthreads();
    }

    // --- epilogue: bias (+GeLU), C/D layout: vgpr r -> m = lhalf*8 + r, n = lmod
    #pragma unroll
    for (int mi = 0; mi < 4; ++mi) {
        #pragma unroll
        for (int ni = 0; ni < 4; ++ni) {
            #pragma unroll
            for (int r = 0; r < 8; ++r) {
                int gm = tileM + wm * 64 + mi * 16 + lhalf * 8 + r;
                int gn = tileN + wn * 64 + ni * 16 + lmod;
                float v = accf[mi][ni][r] + bias[gn];
                if (GELU) v = 0.5f * v * (1.0f + erff(v * 0.70710678118654752f));
                if (OUT_BF16) ((bf16_t*)Cout)[(size_t)gm * N + gn] = (bf16_t)v;
                else          ((float*) Cout)[(size_t)gm * N + gn] = v;
            }
        }
    }
}

// ---------------------------------------------------------------------------
// Launch: fold LoRA into per-expert bf16 weights, bf16-ize x, then 2 WMMA GEMMs.
// ws layout: [x_bf16 | W1m(E,HID,IN) | W2m(E,OUT,HID) | h(M,HID) bf16]  ~334 MB
// ---------------------------------------------------------------------------
extern "C" void kernel_launch(void* const* d_in, const int* in_sizes, int n_in,
                              void* d_out, int out_size, void* d_ws, size_t ws_size,
                              hipStream_t stream)
{
    const float* x     = (const float*)d_in[0];
    const float* w1    = (const float*)d_in[1];
    const float* bias1 = (const float*)d_in[2];
    const float* a1    = (const float*)d_in[3];
    const float* b1    = (const float*)d_in[4];
    const float* w2    = (const float*)d_in[5];
    const float* bias2 = (const float*)d_in[6];
    const float* a2    = (const float*)d_in[7];
    const float* b2    = (const float*)d_in[8];
    const long long* eidx = (const long long*)d_in[9];   // int64 in reference
    const long long* csz  = (const long long*)d_in[10];  // int64 in reference
    const int n_chunks = in_sizes[10];
    float* out = (float*)d_out;

    char* ws = (char*)d_ws;
    size_t off = 0;
    bf16_t* xbf = (bf16_t*)(ws + off); off += (size_t)Mq * INq * sizeof(bf16_t);
    bf16_t* w1m = (bf16_t*)(ws + off); off += (size_t)Eq * HIDq * INq * sizeof(bf16_t);
    bf16_t* w2m = (bf16_t*)(ws + off); off += (size_t)Eq * OUTq * HIDq * sizeof(bf16_t);
    bf16_t* h   = (bf16_t*)(ws + off); off += (size_t)Mq * HIDq * sizeof(bf16_t);
    (void)ws_size; (void)n_in; (void)out_size;

    // 1) x -> bf16
    {
        size_t n4 = (size_t)Mq * INq / 4;
        int blocks = (int)((n4 + 255) / 256);
        f32_to_bf16_kernel<<<blocks, 256, 0, stream>>>(x, xbf, n4);
    }
    // 2) merged per-expert weights (exact LoRA fold), bf16
    merge_lora_kernel<<<dim3(HIDq, Eq), 256, 0, stream>>>(w1, a1, b1, w1m, HIDq, INq, Rq);
    merge_lora_kernel<<<dim3(OUTq, Eq), 256, 0, stream>>>(w2, a2, b2, w2m, OUTq, HIDq, Rq);

    // 3) h = gelu(x @ W1m[e]^T + bias1)   (M x HID), bf16 out
    gemm_bf16_wmma<true, true><<<dim3(HIDq / BN, Mq / BM), 256, 0, stream>>>(
        xbf, w1m, bias1, h, Mq, HIDq, INq, Sq, eidx, csz, n_chunks);

    // 4) y = h @ W2m[e]^T + bias2         (M x OUT), f32 out
    gemm_bf16_wmma<false, false><<<dim3(OUTq / BN, Mq / BM), 256, 0, stream>>>(
        h, w2m, bias2, out, Mq, OUTq, HIDq, Sq, eidx, csz, n_chunks);
}
